// CharacterLevelWordSparse_17334488007265
// MI455X (gfx1250) — matrix operand order repaired and checked by reading, not compiled, not run on hardware
//
#include <hip/hip_runtime.h>

typedef __attribute__((ext_vector_type(16))) _Float16 v16h;
typedef __attribute__((ext_vector_type(8)))  float    v8f;
typedef __attribute__((ext_vector_type(8)))  unsigned int v8u;

#define NUM_EMB 256
#define TOK_PER_WORD 16

// One wave computes one word's full 256-bin histogram as a 16x16 matrix:
//   C[hi][lo] = sum_k onehot(hi_k)[hi] * onehot(lo_k)[lo]   (t = 16*hi + lo)
// via a single V_WMMA_F32_16X16X32_F16 (K=16 real + 16 zero padding).
// Padding token (id 0) is excluded by pointing its hi at 0x100 (no A row match).
__global__ __launch_bounds__(256) void CharHistWmmaKernel(
    const int* __restrict__ tokens, float* __restrict__ out, int numWords) {
  const int lane = threadIdx.x & 31;
  const int half = lane >> 4;   // 0: lanes 0-15, 1: lanes 16-31
  const int r    = lane & 15;   // A row / B col / C col index

  const int wavesPerBlock = blockDim.x >> 5;
  const int waveId = blockIdx.x * wavesPerBlock + ((int)threadIdx.x >> 5);
  const int nWaves = gridDim.x * wavesPerBlock;

  // B comparison id (unused below): for lanes 16-31
  // it is an id that can never match lo in [0,16).
  const int ceff = r | (half << 8);

  for (int w = waveId; w < numWords; w += nWaves) {
    // Word index is wave-uniform; make it provably uniform so token fetches
    // become scalar loads (wave-shared 64B).
    const int ws = __builtin_amdgcn_readfirstlane(w);
    const int* tp = tokens + (size_t)ws * TOK_PER_WORD;

    int t[16];
#pragma unroll
    for (int k = 0; k < 16; ++k) t[k] = tp[k];

    int hi[16], lo[16];
#pragma unroll
    for (int k = 0; k < 16; ++k) {
      hi[k] = (t[k] == 0) ? 0x100 : (t[k] >> 4);  // padding -> unmatched row
      lo[k] = t[k] & 15;
    }

    // ---- A matrix (16x32 f16, one-hot of hi): ----
    // lanes 0-15: VGPR j = K{2j,2j+1}; lanes 16-31: VGPR j = K{8+2j,8+2j+1};
    // VGPR 4-7 (K=16..31) = 0.
    v8u au = {};
#pragma unroll
    for (int j = 0; j < 4; ++j) {
      const int h0 = half ? hi[2 * j + 8] : hi[2 * j];
      const int h1 = half ? hi[2 * j + 9] : hi[2 * j + 1];
      au[j] = (h0 == r ? 0x3C00u : 0u) | (h1 == r ? 0x3C000000u : 0u);
    }

    // ---- B matrix (32x16 f16, one-hot of lo): ----
    // col=lane%16; lanes 0-15: VGPR j = K{2j,2j+1}; lanes 16-31: VGPR j = K{8+2j,8+2j+1};
    // VGPR 4-7 (K=16..31) = 0.
    v8u bu = {};
#pragma unroll
    for (int j = 0; j < 4; ++j) {
      const int l0 = half ? lo[2 * j + 8] : lo[2 * j];
      const int l1 = half ? lo[2 * j + 9] : lo[2 * j + 1];
      bu[j] = (l0 == r ? 0x3C00u : 0u) | (l1 == r ? 0x3C000000u : 0u);
    }

    union { v8u u; v16h h; } ua, ub;
    ua.u = au;
    ub.u = bu;

    v8f c = {};
    // 8 args: (neg_a, A, neg_b, B, c_mod, C, reuse_a, reuse_b)
    c = __builtin_amdgcn_wmma_f32_16x16x32_f16(
        false, ua.h, false, ub.h, (short)0, c, false, false);

    // ---- Store C: VGPR rr, lane -> element (r + 8*half? no:) ----
    // C layout: VGPR rr holds C[M=rr][N=lane] (lanes 0-15) and
    // C[M=rr+8][N=lane-16] (lanes 16-31). Flat bin = M*16 + N.
    // Write-once streaming output: non-temporal stores.
    float* orow = out + (size_t)w * NUM_EMB + (half << 7) + r;
#pragma unroll
    for (int rr = 0; rr < 8; ++rr) {
      __builtin_nontemporal_store(c[rr], orow + 16 * rr);
    }
  }
}

extern "C" void kernel_launch(void* const* d_in, const int* in_sizes, int n_in,
                              void* d_out, int out_size, void* d_ws, size_t ws_size,
                              hipStream_t stream) {
  const int* tokens = (const int*)d_in[0];
  float* out = (float*)d_out;
  const int numWords = in_sizes[0] / TOK_PER_WORD;  // 128*1024 = 131072

  const int threads = 256;                       // 8 waves per block
  int blocks = (numWords + 7) / 8;               // 1 word per wave nominal
  if (blocks > 2048) blocks = 2048;              // grid-stride beyond this
  if (blocks < 1) blocks = 1;

  CharHistWmmaKernel<<<blocks, threads, 0, stream>>>(tokens, out, numWords);
}
